// OutLayer_11673721110551
// MI455X (gfx1250) — matrix operand order repaired and checked by reading, not compiled, not run on hardware
//
#include <hip/hip_runtime.h>
#include <hip/hip_bf16.h>

// Problem constants (match reference)
#define BQ 4
#define NQ 2048
#define KQ 32
#define HQ 128
#define FEAT 512          // H + NUM_IN = 128 + 384
#define SCALE_INV (1.0f / 30.0f)
#define LN_EPS 1e-5f

typedef __attribute__((ext_vector_type(16))) __bf16 bf16x16;
typedef __attribute__((ext_vector_type(8)))  float  f32x8;

union AFrag {
    bf16x16 v;
    uint4   q[2];
};

__device__ __forceinline__ unsigned short f2bf(float f) {
    unsigned int u = __float_as_uint(f);
    u += 0x7FFFu + ((u >> 16) & 1u);          // round-to-nearest-even
    return (unsigned short)(u >> 16);
}

__device__ __forceinline__ float gelu_exact(float x) {
    return 0.5f * x * (1.0f + erff(x * 0.70710678118654752f));
}

// ---------------------------------------------------------------------------
// Prologue: convert W1 (512x128) / W2 (128x128) f32 -> bf16, pre-swizzled into
// the WMMA B-fragment order so the main kernel loads 32 contiguous bytes/lane:
//   w_s[((kstep*2 + khalf)*128 + col)*16 + j] = bf16(W[kstep*32 + khalf*16 + j][col])
// ---------------------------------------------------------------------------
__global__ void swizzle_weights(const float* __restrict__ W1,
                                const float* __restrict__ W2,
                                unsigned short* __restrict__ w1s,
                                unsigned short* __restrict__ w2s) {
    int i = blockIdx.x * blockDim.x + threadIdx.x;
    if (i < FEAT * HQ) {
        int j   = i & 15;
        int col = (i >> 4) & 127;
        int kh  = (i >> 11) & 1;
        int kk  = i >> 12;
        w1s[i] = f2bf(W1[(kk * 32 + kh * 16 + j) * HQ + col]);
    }
    if (i < HQ * HQ) {
        int j   = i & 15;
        int col = (i >> 4) & 127;
        int kh  = (i >> 11) & 1;
        int kk  = i >> 12;
        w2s[i] = f2bf(W2[(kk * 32 + kh * 16 + j) * HQ + col]);
    }
}

// ---------------------------------------------------------------------------
// Main kernel: one workgroup per node (b,n). 8 waves, each owning a 16-wide
// column slice of the hidden/output dimension.
// ---------------------------------------------------------------------------
__global__ __launch_bounds__(256)
void edge_mlp_kernel(const float* __restrict__ hS, const float* __restrict__ hV,
                     const float* __restrict__ hE, const int* __restrict__ Eidx,
                     const float* __restrict__ mask,
                     const float* __restrict__ b1, const float* __restrict__ b2,
                     const float* __restrict__ lng, const float* __restrict__ lnb,
                     const unsigned short* __restrict__ w1s,
                     const unsigned short* __restrict__ w2s,
                     float* __restrict__ out) {
    __shared__ unsigned short ldsA[KQ * FEAT];   // 32 KB: 32x512 edge features (bf16)
    __shared__ unsigned short ldsH[KQ * HQ];     //  8 KB: 32x128 hidden (bf16)
    __shared__ float ldsDh[HQ];
    __shared__ float ldsX[HQ];
    __shared__ float ldsStats[2];

    const int node = blockIdx.x;                 // b*N + n
    const int tid  = threadIdx.x;
    const int bBase = (node / NQ) * NQ;          // b*N

    // ---- Stage 32x512 edge-feature matrix into LDS as bf16 (gather + mask) ----
    {
        const int k   = tid >> 3;                // edge 0..31
        const int seg = tid & 7;                 // 64-feature segment
        const int c0  = seg * 64;
        const int eoff = node * KQ + k;
        const float m  = mask[eoff];
        const int idx  = Eidx[eoff];
        const float* src;
        float scale;
        if (seg < 2)      { src = hV + (long long)node * HQ + c0;            scale = 1.0f; }
        else if (seg < 4) { src = hE + (long long)eoff * 128 + (c0 - 128);   scale = m; }
        else if (seg < 6) { src = hS + (long long)(bBase + idx) * 128 + (c0 - 256); scale = m; }
        else              { src = hV + (long long)(bBase + idx) * 128 + (c0 - 384); scale = m; }
        unsigned short* dst = &ldsA[k * FEAT + c0];
        const float4* s4 = (const float4*)src;
#pragma unroll
        for (int i = 0; i < 16; ++i) {
            float4 v = s4[i];
            uint2 p;
            p.x = (unsigned)f2bf(v.x * scale) | ((unsigned)f2bf(v.y * scale) << 16);
            p.y = (unsigned)f2bf(v.z * scale) | ((unsigned)f2bf(v.w * scale) << 16);
            *(uint2*)(dst + i * 4) = p;
        }
    }
    __syncthreads();

    const int lane = tid & 31;
    const int wid  = tid >> 5;        // wave id -> column tile
    const int lrow = lane & 15;
    const int lhk  = lane >> 4;       // K-half select (this code's contiguous order: k = 16*lhk + s)
    const int n0   = wid * 16;
    const int col  = n0 + lrow;

    // ---- GEMM1: (32x512) x (512x128), bf16 WMMA, f32 accumulate ----
    f32x8 acc0 = {};
    f32x8 acc1 = {};
#pragma unroll
    for (int kk = 0; kk < 16; ++kk) {
        AFrag bfr;
        const uint4* bp = (const uint4*)(w1s + (size_t)(((kk * 2 + lhk) * HQ) + col) * 16);
        bfr.q[0] = bp[0];
        bfr.q[1] = bp[1];
        const int abase = kk * 32 + lhk * 16;
        AFrag a0, a1;
        a0.q[0] = *(const uint4*)(ldsA + lrow * FEAT + abase);
        a0.q[1] = *(const uint4*)(ldsA + lrow * FEAT + abase + 8);
        a1.q[0] = *(const uint4*)(ldsA + (lrow + 16) * FEAT + abase);
        a1.q[1] = *(const uint4*)(ldsA + (lrow + 16) * FEAT + abase + 8);
        acc0 = __builtin_amdgcn_wmma_f32_16x16x32_bf16(false, a0.v, false, bfr.v,
                                                       (short)0, acc0, false, false);
        acc1 = __builtin_amdgcn_wmma_f32_16x16x32_bf16(false, a1.v, false, bfr.v,
                                                       (short)0, acc1, false, false);
    }

    // ---- bias + exact GELU, restage hidden as bf16 for GEMM2 ----
    {
        const float bias1 = b1[col];
#pragma unroll
        for (int r = 0; r < 8; ++r) {
            const int m0 = r + 8 * lhk;                    // C/D layout: M = r + 8*(lane>=16)
            ldsH[m0 * HQ + col]        = f2bf(gelu_exact(acc0[r] + bias1));
            ldsH[(m0 + 16) * HQ + col] = f2bf(gelu_exact(acc1[r] + bias1));
        }
    }
    __syncthreads();

    // ---- GEMM2: (32x128) x (128x128) ----
    f32x8 macc0 = {};
    f32x8 macc1 = {};
#pragma unroll
    for (int kk = 0; kk < 4; ++kk) {
        AFrag bfr;
        const uint4* bp = (const uint4*)(w2s + (size_t)(((kk * 2 + lhk) * HQ) + col) * 16);
        bfr.q[0] = bp[0];
        bfr.q[1] = bp[1];
        const int abase = kk * 32 + lhk * 16;
        AFrag a0, a1;
        a0.q[0] = *(const uint4*)(ldsH + lrow * HQ + abase);
        a0.q[1] = *(const uint4*)(ldsH + lrow * HQ + abase + 8);
        a1.q[0] = *(const uint4*)(ldsH + (lrow + 16) * HQ + abase);
        a1.q[1] = *(const uint4*)(ldsH + (lrow + 16) * HQ + abase + 8);
        macc0 = __builtin_amdgcn_wmma_f32_16x16x32_bf16(false, a0.v, false, bfr.v,
                                                        (short)0, macc0, false, false);
        macc1 = __builtin_amdgcn_wmma_f32_16x16x32_bf16(false, a1.v, false, bfr.v,
                                                        (short)0, macc1, false, false);
    }

    // ---- sum h_message over K=32 edges per column, + bias, / SCALE ----
    {
        float ps = 0.0f;
#pragma unroll
        for (int r = 0; r < 8; ++r) ps += macc0[r] + macc1[r];   // 16 of 32 rows
        ps += __shfl_xor(ps, 16, 32);                            // other 16 rows (wave32)
        if (lane < 16)
            ldsDh[col] = (ps + 32.0f * b2[col]) * SCALE_INV;
    }
    __syncthreads();

    // ---- residual + LayerNorm ----
    if (tid < HQ) ldsX[tid] = hV[(long long)node * HQ + tid] + ldsDh[tid];
    __syncthreads();

    if (wid == 0) {
        float s = 0.0f, ss = 0.0f;
#pragma unroll
        for (int i = 0; i < 4; ++i) {
            float x = ldsX[lane + 32 * i];
            s  += x;
            ss += x * x;
        }
#pragma unroll
        for (int off = 16; off >= 1; off >>= 1) {
            s  += __shfl_xor(s, off, 32);
            ss += __shfl_xor(ss, off, 32);
        }
        if (lane == 0) {
            float mean = s * (1.0f / 128.0f);
            float var  = ss * (1.0f / 128.0f) - mean * mean;
            ldsStats[0] = mean;
            ldsStats[1] = rsqrtf(var + LN_EPS);
        }
    }
    __syncthreads();

    if (tid < HQ) {
        float x = (ldsX[tid] - ldsStats[0]) * ldsStats[1];
        out[(long long)node * HQ + tid] = x * lng[tid] + lnb[tid];
    }
}

// ---------------------------------------------------------------------------
extern "C" void kernel_launch(void* const* d_in, const int* in_sizes, int n_in,
                              void* d_out, int out_size, void* d_ws, size_t ws_size,
                              hipStream_t stream) {
    const float* hS   = (const float*)d_in[0];
    const float* hV   = (const float*)d_in[1];
    const float* hE   = (const float*)d_in[2];
    const int*   Eidx = (const int*)  d_in[3];
    const float* mask = (const float*)d_in[4];
    const float* W1   = (const float*)d_in[5];
    const float* b1   = (const float*)d_in[6];
    const float* W2   = (const float*)d_in[7];
    const float* b2   = (const float*)d_in[8];
    const float* lng  = (const float*)d_in[9];
    const float* lnb  = (const float*)d_in[10];

    unsigned short* w1s = (unsigned short*)d_ws;            // 512*128 bf16 = 128 KB
    unsigned short* w2s = w1s + FEAT * HQ;                  // 128*128 bf16 =  32 KB

    // Weight convert+swizzle (cheap; runs every call -> deterministic/capture-safe)
    hipLaunchKernelGGL(swizzle_weights, dim3(256), dim3(256), 0, stream,
                       W1, W2, w1s, w2s);

    // One workgroup per node
    hipLaunchKernelGGL(edge_mlp_kernel, dim3(BQ * NQ), dim3(256), 0, stream,
                       hS, hV, hE, Eidx, mask, b1, b2, lng, lnb, w1s, w2s,
                       (float*)d_out);
}
